// BiMambaBlock_55903294324787
// MI455X (gfx1250) — matrix operand order repaired and checked by reading, not compiled, not run on hardware
//
#include <hip/hip_runtime.h>

// ---------------------------------------------------------------------------
// Types for CDNA5 WMMA (gfx1250, wave32)
// ---------------------------------------------------------------------------
typedef __attribute__((ext_vector_type(16))) __bf16  v16bf;
typedef __attribute__((ext_vector_type(8)))  float   v8f;
typedef __attribute__((ext_vector_type(4)))  unsigned int v4u;

union FragBF { v16bf v; v4u q[2]; unsigned short u[16]; };
union FragF  { v8f  v; float f[8]; };

__device__ __forceinline__ unsigned short f2bf(float f) {
    union { float f; unsigned int u; } v; v.f = f;
    unsigned int r = v.u + 0x7FFFu + ((v.u >> 16) & 1u);   // round-to-nearest-even
    return (unsigned short)(r >> 16);
}

// LDS byte offset of a __shared__ object (low 32 bits of the flat address are
// the LDS aperture offset on gfx1250)
__device__ __forceinline__ unsigned lds_off(const void* p) {
    return (unsigned)(unsigned long long)p;
}

// Async copy 16B global -> LDS (ASYNCcnt); portable via inline asm
__device__ __forceinline__ void async_g2l_b128(unsigned ldsaddr, const void* gaddr) {
    asm volatile("global_load_async_to_lds_b128 %0, %1, off"
                 :: "v"(ldsaddr), "v"(gaddr) : "memory");
}
__device__ __forceinline__ void wait_asynccnt0() {
    asm volatile("s_wait_asynccnt 0x0" ::: "memory");
}

// ---------------------------------------------------------------------------
// Problem constants (B=2, L=1024, Dm=1024, Di=2048, N=16, R=64, K=4)
// ---------------------------------------------------------------------------
#define BL   2048        // B*L rows
#define DM   1024
#define DI   2048
#define SN   16
#define RR   64
#define XPN  96          // R + 2N

// ---------------------------------------------------------------------------
// fp32 (KxN) -> bf16 TRANSPOSED (NxK) weight convert
// ---------------------------------------------------------------------------
__global__ void cvt_t_kernel(const float* __restrict__ src,
                             unsigned short* __restrict__ dst, int K, int N) {
    long total = (long)K * N;
    long i = (long)blockIdx.x * blockDim.x + threadIdx.x;
    long st = (long)gridDim.x * blockDim.x;
    for (; i < total; i += st) {
        int k = (int)(i / N), n = (int)(i - (long)k * N);
        dst[(size_t)n * K + k] = f2bf(src[i]);
    }
}

// Pack [Wg | Wv] -> transposed (2048n x 2048k) bf16, and [bg | bv] -> bgv
__global__ void pack_gv_kernel(const float* __restrict__ Wg, const float* __restrict__ Wv,
                               const float* __restrict__ bg, const float* __restrict__ bv,
                               unsigned short* __restrict__ Wgvt, float* __restrict__ bgvb) {
    long i = (long)blockIdx.x * blockDim.x + threadIdx.x;
    if (i >= (long)2048 * 2048) return;
    int n = (int)(i >> 11), k = (int)(i & 2047);
    float w = (n < 1024) ? Wg[(size_t)k * 1024 + n] : Wv[(size_t)k * 1024 + (n - 1024)];
    Wgvt[i] = f2bf(w);
    if (i < 2048) bgvb[i] = (i < 1024) ? bg[i] : bv[i - 1024];
}

// ---------------------------------------------------------------------------
// LayerNorm: one block per row; writes bf16 xn for BOTH directions
// ---------------------------------------------------------------------------
__global__ __launch_bounds__(256) void ln_kernel(const float* __restrict__ x,
                                                 const float* __restrict__ lnw,
                                                 const float* __restrict__ lnb,
                                                 unsigned short* __restrict__ xn0,
                                                 unsigned short* __restrict__ xn1) {
    int row = blockIdx.x;
    const float* xr = x + (size_t)row * DM;
    __shared__ float red[256];
    float s = 0.f;
    for (int c = threadIdx.x; c < DM; c += 256) s += xr[c];
    red[threadIdx.x] = s; __syncthreads();
    for (int st = 128; st > 0; st >>= 1) {
        if (threadIdx.x < st) red[threadIdx.x] += red[threadIdx.x + st];
        __syncthreads();
    }
    float mu = red[0] * (1.f / DM);
    __syncthreads();
    float v = 0.f;
    for (int c = threadIdx.x; c < DM; c += 256) { float d = xr[c] - mu; v += d * d; }
    red[threadIdx.x] = v; __syncthreads();
    for (int st = 128; st > 0; st >>= 1) {
        if (threadIdx.x < st) red[threadIdx.x] += red[threadIdx.x + st];
        __syncthreads();
    }
    float rs = rsqrtf(red[0] * (1.f / DM) + 1e-5f);
    for (int c = threadIdx.x; c < DM; c += 256) {
        float nv = (xr[c] - mu) * rs;
        xn0[(size_t)row * DM + c] = f2bf(nv * lnw[c]      + lnb[c]);
        xn1[(size_t)row * DM + c] = f2bf(nv * lnw[DM + c] + lnb[DM + c]);
    }
}

// ---------------------------------------------------------------------------
// Generic bf16 WMMA GEMM:  C(MxN) = epi( A(MxK,bf16) @ Wt(NxK,bf16)^T + bias )
// Wt PRE-TRANSPOSED (N-major). Staging uses GLOBAL_LOAD_ASYNC_TO_LDS_B128
// (ASYNCcnt) so tile movement bypasses VGPRs entirely.
// Workgroup: 256 thr = 8 waves, 128x128 tile, K-step 64; wave = 64x32.
// EPI: 0 = +bias store   1 = softplus(+bias)   2 = +bias +res, also bf16->aux
//      3 = store, cols<64 also bf16->aux (dt slice)
// ---------------------------------------------------------------------------
template <int EPI>
__global__ __launch_bounds__(256) void gemm_bf16(
    const unsigned short* __restrict__ A, int lda,
    const unsigned short* __restrict__ Wt, int ldwt,
    float* __restrict__ C, int ldc,
    const float* __restrict__ bias,
    const float* __restrict__ res, int ldres,
    unsigned short* __restrict__ auxb, int ldaux, int auxColOff,
    int M, int N, int K) {
    __shared__ __align__(16) unsigned short As[128 * 64];   // [m][k]
    __shared__ __align__(16) unsigned short Ws[128 * 64];   // [n][k]
    const int tid  = threadIdx.x;
    const int lane = tid & 31;
    const int wave = tid >> 5;
    const int mTile = blockIdx.y * 128;
    const int nTile = blockIdx.x * 128;
    const int waveM = (wave & 1) * 64;
    const int waveN = (wave >> 1) * 32;
    const int h    = lane >> 4;      // lane half
    const int ln15 = lane & 15;

    FragF acc[4][2];
#pragma unroll
    for (int i = 0; i < 4; ++i)
#pragma unroll
        for (int j = 0; j < 2; ++j)
#pragma unroll
            for (int r = 0; r < 8; ++r) acc[i][j].f[r] = 0.f;

    // staging coords: each thread moves one 32-ushort (64B) run of one row
    const int sr = tid >> 1;              // row 0..127
    const int sc = (tid & 1) * 32;        // col 0 or 32
    const bool wInB = (nTile + sr) < N;   // W row in-bounds (N=96 tail case)

    const unsigned short* gpA = A  + (size_t)(mTile + sr) * lda  + sc;
    const unsigned short* gpW = Wt + (size_t)(nTile + sr) * ldwt + sc;
    const unsigned ldsA = lds_off(&As[sr * 64 + sc]);
    const unsigned ldsW = lds_off(&Ws[sr * 64 + sc]);

    for (int k0 = 0; k0 < K; k0 += 64) {
        // ---- async-stage A tile (128 x 64): 4 x b128 per thread, no VGPR data ----
        {
            const unsigned short* g = gpA + k0;
#pragma unroll
            for (int q = 0; q < 4; ++q)
                async_g2l_b128(ldsA + q * 16, g + q * 8);
        }
        // ---- async-stage Wt tile (128 x 64); OOB rows (N tail) zero-filled ----
        if (wInB) {
            const unsigned short* g = gpW + k0;
#pragma unroll
            for (int q = 0; q < 4; ++q)
                async_g2l_b128(ldsW + q * 16, g + q * 8);
        } else {
            v4u z; z.x = z.y = z.z = z.w = 0u;
#pragma unroll
            for (int q = 0; q < 4; ++q)
                *(v4u*)&Ws[sr * 64 + sc + q * 8] = z;
        }
        if (k0 + 64 < K) {
            __builtin_prefetch(gpA + k0 + 64, 0, 1);   // global_prefetch_b8
            __builtin_prefetch(gpW + k0 + 64, 0, 1);
        }
        wait_asynccnt0();
        __syncthreads();
        // ---- two K=32 sub-steps per stage ----
#pragma unroll
        for (int ks = 0; ks < 64; ks += 32) {
            FragBF a[4], b[2];
#pragma unroll
            for (int i = 0; i < 4; ++i) {
                int m = waveM + i * 16 + ln15;
                a[i].q[0] = *(const v4u*)&As[m * 64 + ks + h * 8];        // K=h*8..
                a[i].q[1] = *(const v4u*)&As[m * 64 + ks + 16 + h * 8];   // K=16+h*8..
            }
#pragma unroll
            for (int j = 0; j < 2; ++j) {
                int n = waveN + j * 16 + ln15;
                b[j].q[0] = *(const v4u*)&Ws[n * 64 + ks + h * 8];        // K=h*8..
                b[j].q[1] = *(const v4u*)&Ws[n * 64 + ks + 16 + h * 8];
            }
#pragma unroll
            for (int i = 0; i < 4; ++i)
#pragma unroll
                for (int j = 0; j < 2; ++j)
                    acc[i][j].v = __builtin_amdgcn_wmma_f32_16x16x32_bf16(
                        false, a[i].v, false, b[j].v, (short)0, acc[i][j].v, false, false);
        }
        __syncthreads();
    }

    // ---- epilogue: D element (m,n) at lane=(m>>3)*16+n, vgpr r=m&7 ----
#pragma unroll
    for (int i = 0; i < 4; ++i)
#pragma unroll
        for (int j = 0; j < 2; ++j) {
            int ncol = nTile + waveN + j * 16 + ln15;
            if (ncol >= N) continue;
            float bvv = bias ? bias[ncol] : 0.f;
#pragma unroll
            for (int r = 0; r < 8; ++r) {
                int mrow = mTile + waveM + i * 16 + h * 8 + r;
                float v = acc[i][j].f[r] + bvv;
                if (EPI == 1) v = (v > 20.f) ? v : logf(1.f + __expf(v));   // softplus
                if (EPI == 2) v += res[(size_t)mrow * ldres + ncol];
                C[(size_t)mrow * ldc + ncol] = v;
                if (EPI == 2)
                    auxb[(size_t)mrow * ldaux + auxColOff + ncol] = f2bf(v);
                if (EPI == 3 && ncol < RR)
                    auxb[(size_t)mrow * ldaux + ncol] = f2bf(v);
            }
        }
}

// ---------------------------------------------------------------------------
// Causal (dir=0) / anti-causal (dir=1) depthwise conv K=4 + SiLU
// xp = xz[..., :DI]; writes fp32 + bf16 copies
// ---------------------------------------------------------------------------
__global__ void conv_silu_kernel(const float* __restrict__ xz,
                                 const float* __restrict__ cw,
                                 const float* __restrict__ cb,
                                 float* __restrict__ xpf,
                                 unsigned short* __restrict__ xpb, int dir) {
    long id = (long)blockIdx.x * 256 + threadIdx.x;
    if (id >= (long)BL * DI) return;
    int c = (int)(id & (DI - 1));
    int t = (int)((id >> 11) & 1023);
    int b = (int)(id >> 21);
    float acc = cb[c];
#pragma unroll
    for (int j = 0; j < 4; ++j) {
        int tt = dir ? (t + 3 - j) : (t - 3 + j);
        if (tt >= 0 && tt < 1024)
            acc += cw[c * 4 + j] * xz[((size_t)(b * 1024 + tt)) * (2 * DI) + c];
    }
    float s = acc / (1.f + __expf(-acc));
    size_t o = (size_t)(b * 1024 + t) * DI + c;
    xpf[o] = s;
    xpb[o] = f2bf(s);
}

// ---------------------------------------------------------------------------
// Fused selective-scan. One lane = one (batch, channel); 16 states in VGPRs.
// dA/dBx recomputed on the fly; epilogue (+xp*D)*silu(z) fused; emits bf16 y.
// grid = 16 blocks x 256 thr  (2 batches * 2048 channels)
// ---------------------------------------------------------------------------
__global__ __launch_bounds__(256) void scan_kernel(const float* __restrict__ delta,
                                                   const float* __restrict__ dbl,
                                                   const float* __restrict__ xpf,
                                                   const float* __restrict__ xz,
                                                   const float* __restrict__ A_log,
                                                   const float* __restrict__ Dp,
                                                   unsigned short* __restrict__ ybf,
                                                   int dir) {
    int b = blockIdx.x >> 3;
    int c = ((blockIdx.x & 7) << 8) + threadIdx.x;
    __shared__ float BmS[SN], CmS[SN];
    float Acf[SN], h[SN];
#pragma unroll
    for (int n = 0; n < SN; ++n) {
        Acf[n] = -__expf(A_log[c * SN + n]);
        h[n] = 0.f;
    }
    float dcoef = Dp[c];
    for (int s = 0; s < 1024; ++s) {
        int t = dir ? (1023 - s) : s;
        int row = b * 1024 + t;
        __syncthreads();
        if (threadIdx.x < SN)
            BmS[threadIdx.x] = dbl[(size_t)row * XPN + RR + threadIdx.x];
        else if (threadIdx.x < 32)
            CmS[threadIdx.x - SN] = dbl[(size_t)row * XPN + RR + threadIdx.x];
        __syncthreads();
        float dlt = delta[(size_t)row * DI + c];
        float xv  = xpf[(size_t)row * DI + c];
        float dx  = dlt * xv;
        float y = 0.f;
#pragma unroll
        for (int n = 0; n < SN; ++n) {
            float e = __expf(dlt * Acf[n]);
            h[n] = e * h[n] + dx * BmS[n];
            y += h[n] * CmS[n];
        }
        float z = xz[(size_t)row * (2 * DI) + DI + c];
        float sil = z / (1.f + __expf(-z));
        float yv = (y + xv * dcoef) * sil;
        ybf[(size_t)row * DI + c] = f2bf(yv);
    }
}

// ---------------------------------------------------------------------------
// Final gated combine: out = (g*v + (1-g)*fwd + (1-g)*bwd)/2,  g=sigmoid(gv_g)
// ---------------------------------------------------------------------------
__global__ void final_kernel(const float* __restrict__ gv,
                             const float* __restrict__ mo0,
                             const float* __restrict__ mo1,
                             float* __restrict__ out) {
    long id = (long)blockIdx.x * 256 + threadIdx.x;
    if (id >= (long)BL * DM) return;
    int row = (int)(id >> 10), c = (int)(id & 1023);
    float g = 1.f / (1.f + __expf(-gv[(size_t)row * 2048 + c]));
    float v = gv[(size_t)row * 2048 + 1024 + c];
    out[id] = 0.5f * (g * v + (1.f - g) * (mo0[id] + mo1[id]));
}

// ---------------------------------------------------------------------------
// Host orchestration
// ---------------------------------------------------------------------------
extern "C" void kernel_launch(void* const* d_in, const int* in_sizes, int n_in,
                              void* d_out, int out_size, void* d_ws, size_t ws_size,
                              hipStream_t stream) {
    const float* x      = (const float*)d_in[0];
    const float* ln_w   = (const float*)d_in[1];
    const float* ln_b   = (const float*)d_in[2];
    const float* W_in   = (const float*)d_in[3];
    const float* b_in   = (const float*)d_in[4];
    const float* conv_w = (const float*)d_in[5];
    const float* conv_b = (const float*)d_in[6];
    const float* W_xp   = (const float*)d_in[7];
    const float* W_dt   = (const float*)d_in[8];
    const float* b_dt   = (const float*)d_in[9];
    const float* A_log  = (const float*)d_in[10];
    const float* Dvec   = (const float*)d_in[11];
    const float* W_out  = (const float*)d_in[12];
    const float* b_out  = (const float*)d_in[13];
    const float* Wg     = (const float*)d_in[14];
    const float* bg     = (const float*)d_in[15];
    const float* Wv     = (const float*)d_in[16];
    const float* bv     = (const float*)d_in[17];
    float* out = (float*)d_out;

    char* wsp = (char*)d_ws;
    size_t off = 0;
    auto alloc = [&](size_t bytes) -> void* {
        void* p = wsp + off;
        off += (bytes + 255) & ~(size_t)255;
        return p;
    };
    // transposed bf16 weights (N-major)
    unsigned short* Winb  = (unsigned short*)alloc(2ull * (2 * DI) * DM * 2);
    unsigned short* Wxpb  = (unsigned short*)alloc(2ull * XPN * DI * 2);
    unsigned short* Wdtb  = (unsigned short*)alloc(2ull * DI * RR * 2);
    unsigned short* Woutb = (unsigned short*)alloc(2ull * DM * DI * 2);
    unsigned short* Wgvb  = (unsigned short*)alloc(2048ull * 2048 * 2);
    float*          bgvb  = (float*)alloc(2048 * 4);
    unsigned short* xn[2], *xpb[2], *dtb[2], *ybf[2];
    float *xzb[2], *xpf[2], *dblb[2], *dltb[2], *mo[2];
    for (int d = 0; d < 2; ++d) {
        xn[d]   = (unsigned short*)alloc((size_t)BL * DM * 2);
        xzb[d]  = (float*)alloc((size_t)BL * 2 * DI * 4);
        xpf[d]  = (float*)alloc((size_t)BL * DI * 4);
        xpb[d]  = (unsigned short*)alloc((size_t)BL * DI * 2);
        dblb[d] = (float*)alloc((size_t)BL * XPN * 4);
        dtb[d]  = (unsigned short*)alloc((size_t)BL * RR * 2);
        dltb[d] = (float*)alloc((size_t)BL * DI * 4);
        ybf[d]  = (unsigned short*)alloc((size_t)BL * DI * 2);
        mo[d]   = (float*)alloc((size_t)BL * DM * 4);
    }
    unsigned short* comb = (unsigned short*)alloc((size_t)BL * 2048 * 2);
    float*          gv   = (float*)alloc((size_t)BL * 2048 * 4);

    // --- weight conversion (to transposed bf16) ---
    auto cvtT = [&](const float* s, unsigned short* dst, int K, int N) {
        long n = (long)K * N;
        int blocks = (int)((n + 255) / 256);
        if (blocks > 16384) blocks = 16384;
        cvt_t_kernel<<<blocks, 256, 0, stream>>>(s, dst, K, N);
    };
    for (int d = 0; d < 2; ++d) {
        cvtT(W_in  + (size_t)d * DM * 2 * DI, Winb  + (size_t)d * (2 * DI) * DM, DM, 2 * DI);
        cvtT(W_xp  + (size_t)d * DI * XPN,    Wxpb  + (size_t)d * XPN * DI,      DI, XPN);
        cvtT(W_dt  + (size_t)d * RR * DI,     Wdtb  + (size_t)d * DI * RR,       RR, DI);
        cvtT(W_out + (size_t)d * DI * DM,     Woutb + (size_t)d * DM * DI,       DI, DM);
    }
    pack_gv_kernel<<<(2048 * 2048) / 256, 256, 0, stream>>>(Wg, Wv, bg, bv, Wgvb, bgvb);

    // --- layernorm (both directions in one pass) ---
    ln_kernel<<<BL, 256, 0, stream>>>(x, ln_w, ln_b, xn[0], xn[1]);

    dim3 blk(256);
    for (int d = 0; d < 2; ++d) {
        // xz = LN(x) @ W_in + b_in            (2048 x 4096, K=1024)
        gemm_bf16<0><<<dim3((2 * DI) / 128, BL / 128), blk, 0, stream>>>(
            xn[d], DM, Winb + (size_t)d * (2 * DI) * DM, DM,
            xzb[d], 2 * DI, b_in + (size_t)d * 2 * DI,
            nullptr, 0, nullptr, 0, 0, BL, 2 * DI, DM);
        // depthwise conv + SiLU
        conv_silu_kernel<<<(BL * DI) / 256, blk, 0, stream>>>(
            xzb[d], conv_w + (size_t)d * DI * 4, conv_b + (size_t)d * DI,
            xpf[d], xpb[d], d);
        // dbl = xp @ W_xproj                  (2048 x 96, K=2048), dt-slice -> bf16
        gemm_bf16<3><<<dim3(1, BL / 128), blk, 0, stream>>>(
            xpb[d], DI, Wxpb + (size_t)d * XPN * DI, DI,
            dblb[d], XPN, nullptr,
            nullptr, 0, dtb[d], RR, 0, BL, XPN, DI);
        // delta = softplus(dt @ W_dt + b_dt)  (2048 x 2048, K=64)
        gemm_bf16<1><<<dim3(DI / 128, BL / 128), blk, 0, stream>>>(
            dtb[d], RR, Wdtb + (size_t)d * DI * RR, RR,
            dltb[d], DI, b_dt + (size_t)d * DI,
            nullptr, 0, nullptr, 0, 0, BL, DI, RR);
        // fused selective scan -> bf16 y
        scan_kernel<<<16, blk, 0, stream>>>(
            dltb[d], dblb[d], xpf[d], xzb[d],
            A_log + (size_t)d * DI * SN, Dvec + (size_t)d * DI, ybf[d], d);
        // mo = x + y @ W_out + b_out  (also write bf16 into comb[:, d*1024:])
        gemm_bf16<2><<<dim3(DM / 128, BL / 128), blk, 0, stream>>>(
            ybf[d], DI, Woutb + (size_t)d * DM * DI, DI,
            mo[d], DM, b_out + (size_t)d * DM,
            x, DM, comb, 2048, d * 1024, BL, DM, DI);
    }

    // gv = comb @ [Wg|Wv] + [bg|bv]           (2048 x 2048, K=2048)
    gemm_bf16<0><<<dim3(2048 / 128, BL / 128), blk, 0, stream>>>(
        comb, 2048, Wgvb, 2048, gv, 2048, bgvb,
        nullptr, 0, nullptr, 0, 0, BL, 2048, 2048);

    // final sigmoid-gated combine
    final_kernel<<<(BL * DM) / 256, blk, 0, stream>>>(gv, mo[0], mo[1], out);
}
